// GATModel_22256520528146
// MI455X (gfx1250) — matrix operand order repaired and checked by reading, not compiled, not run on hardware
//
#include <hip/hip_runtime.h>
#include <hip/hip_bf16.h>
#include <math.h>

// ---- CDNA5 WMMA vector types ----
typedef __attribute__((ext_vector_type(16))) _Float16 v16h;
typedef __attribute__((ext_vector_type(8)))  float    v8f;

#define D  128
#define NH 4
#define HC 32

// -------------------- utility --------------------
__global__ void k_fill(float* __restrict__ p, float v, int n) {
  int i = blockIdx.x * blockDim.x + threadIdx.x;
  if (i < n) p[i] = v;
}

__device__ inline void atomicMaxF(float* a, float v) {
  // sign-split trick: positive floats ordered as ints, negative as reversed uints
  if (v >= 0.0f) atomicMax((int*)a, __float_as_int(v));
  else           atomicMin((unsigned int*)a, __float_as_uint(v));
}

// -------------------- WMMA GEMM: C[nrows,128] = A[nrows,128] @ W[128,128] (+bias) --------------------
// block = 128 threads (4 waves). Each wave computes a 16-row strip across all 128 cols.
__global__ __launch_bounds__(128) void k_gemm128(
    const float* __restrict__ A, const float* __restrict__ W,
    const float* __restrict__ bias, float* __restrict__ Cout, int nrows) {
  __shared__ __align__(16) _Float16 As[64 * 128];   // 16 KB: 64 A rows, row-major f16
  __shared__ __align__(16) _Float16 Bt[128 * 128];  // 32 KB: W transposed, Bt[n][k] = W[k][n]

  const int tid  = threadIdx.x;
  const int row0 = blockIdx.x * 64;

  // stage W transposed as f16
  for (int i = tid; i < 128 * 128; i += 128) {
    int k = i >> 7, j = i & 127;
    Bt[j * 128 + k] = (_Float16)W[i];
  }
  // stage A tile as f16 (zero-pad past nrows)
  for (int i = tid; i < 64 * 128; i += 128) {
    int r = i >> 7, c = i & 127;
    int row = row0 + r;
    As[i] = (row < nrows) ? (_Float16)A[(size_t)row * 128 + c] : (_Float16)0.0f;
  }
  __syncthreads();

  const int lane = tid & 31;
  const int wv   = tid >> 5;
  const int lr   = lane & 15;   // row/col within tile
  const int hi   = lane >> 4;   // lane-group

  v8f acc[8] = {};

  #pragma unroll
  for (int kk = 0; kk < 4; ++kk) {
    // A fragment: lane<16 holds K {0..7,16..23}, lane>=16 holds K {8..15,24..31} (ISA 7.12.2)
    v16h af;
    const _Float16* ap = &As[(wv * 16 + lr) * 128 + kk * 32 + hi * 8];
    ((uint4*)&af)[0] = *(const uint4*)(ap);
    ((uint4*)&af)[1] = *(const uint4*)(ap + 16);
    #pragma unroll
    for (int t = 0; t < 8; ++t) {
      // B fragment from transposed LDS: lane n = lr, K {0..7,16..23} (lanes>=16: K {8..15,24..31})
      v16h bf;
      const _Float16* bp = &Bt[(t * 16 + lr) * 128 + kk * 32 + hi * 8];
      ((uint4*)&bf)[0] = *(const uint4*)(bp);
      ((uint4*)&bf)[1] = *(const uint4*)(bp + 16);
      acc[t] = __builtin_amdgcn_wmma_f32_16x16x32_f16(
          false, af, false, bf, (short)0, acc[t], false, false);
    }
  }

  // C layout: VGPR r -> M = r + 8*hi, N = lr + 16*t
  #pragma unroll
  for (int t = 0; t < 8; ++t) {
    int col = t * 16 + lr;
    float bv = bias ? bias[col] : 0.0f;
    #pragma unroll
    for (int r = 0; r < 8; ++r) {
      int row = row0 + wv * 16 + r + hi * 8;
      if (row < nrows) Cout[(size_t)row * 128 + col] = acc[t][r] + bv;
    }
  }
}

// -------------------- per-node attention scores + init m/z --------------------
__global__ void k_scores(const float* __restrict__ xh,
                         const float* __restrict__ asrc, const float* __restrict__ adst,
                         float* __restrict__ es, float* __restrict__ ed,
                         float* __restrict__ m, float* __restrict__ z, int Nn) {
  int i = blockIdx.x * blockDim.x + threadIdx.x;
  if (i >= Nn * NH) return;
  int n = i >> 2, h = i & 3;
  const float* xp = xh + (size_t)n * D + h * HC;
  float s = 0.0f, d = 0.0f;
  #pragma unroll
  for (int c = 0; c < HC; ++c) {
    float v = xp[c];
    s += v * asrc[h * HC + c];
    d += v * adst[h * HC + c];
  }
  es[i] = s; ed[i] = d;
  m[i] = -3.0e38f; z[i] = 0.0f;
}

// -------------------- edge pass 1: logits + segment max --------------------
__global__ void k_edge1(const int* __restrict__ ei,
                        const float* __restrict__ es, const float* __restrict__ ed,
                        float* __restrict__ e, float* __restrict__ m, int Ee, int Etot) {
  int i = blockIdx.x * blockDim.x + threadIdx.x;
  if (i >= Etot) return;
  int s = (i < Ee) ? ei[i]      : (i - Ee);
  int d = (i < Ee) ? ei[Ee + i] : (i - Ee);
  #pragma unroll
  for (int h = 0; h < NH; ++h) {
    float v = es[s * NH + h] + ed[d * NH + h];
    v = (v > 0.0f) ? v : 0.2f * v;           // leaky_relu(0.2)
    e[(size_t)i * NH + h] = v;
    atomicMaxF(&m[d * NH + h], v);
  }
}

// -------------------- edge pass 2: exp + segment sum --------------------
__global__ void k_edge2(const int* __restrict__ ei, float* __restrict__ e,
                        const float* __restrict__ m, float* __restrict__ z, int Ee, int Etot) {
  int i = blockIdx.x * blockDim.x + threadIdx.x;
  if (i >= Etot) return;
  int d = (i < Ee) ? ei[Ee + i] : (i - Ee);
  #pragma unroll
  for (int h = 0; h < NH; ++h) {
    float v = __expf(e[(size_t)i * NH + h] - m[d * NH + h]);
    e[(size_t)i * NH + h] = v;
    atomicAdd(&z[d * NH + h], v);
  }
}

// -------------------- edge pass 3: weighted scatter-add (wave per edge) --------------------
__global__ __launch_bounds__(256) void k_edge3(
    const int* __restrict__ ei, const float* __restrict__ e, const float* __restrict__ z,
    const float* __restrict__ xh, float* __restrict__ acc, int Ee, int Etot) {
  int gid  = blockIdx.x * 8 + (threadIdx.x >> 5);
  int lane = threadIdx.x & 31;
  if (gid >= Etot) return;
  int s = (gid < Ee) ? ei[gid]      : (gid - Ee);
  int d = (gid < Ee) ? ei[Ee + gid] : (gid - Ee);
  int h = lane >> 3;                                   // lane covers dims [4*lane,4*lane+4)
  float alpha = e[(size_t)gid * NH + h] / z[d * NH + h];
  const float4 v = *(const float4*)(xh + (size_t)s * D + lane * 4);
  float* ap = acc + (size_t)d * D + lane * 4;
  atomicAdd(ap + 0, v.x * alpha);
  atomicAdd(ap + 1, v.y * alpha);
  atomicAdd(ap + 2, v.z * alpha);
  atomicAdd(ap + 3, v.w * alpha);
}

// -------------------- bias + LayerNorm + ReLU + residual (wave per node) --------------------
__global__ __launch_bounds__(256) void k_finalize(
    const float* __restrict__ acc, const float* __restrict__ cb,
    const float* __restrict__ lg, const float* __restrict__ lb,
    float* __restrict__ h, int Nn) {
  int n    = blockIdx.x * 8 + (threadIdx.x >> 5);
  int lane = threadIdx.x & 31;
  if (n >= Nn) return;
  float4 c  = *(const float4*)(acc + (size_t)n * D + lane * 4);
  float4 b4 = *(const float4*)(cb + lane * 4);
  c.x += b4.x; c.y += b4.y; c.z += b4.z; c.w += b4.w;
  float s = c.x + c.y + c.z + c.w;
  float q = c.x * c.x + c.y * c.y + c.z * c.z + c.w * c.w;
  #pragma unroll
  for (int o = 16; o > 0; o >>= 1) { s += __shfl_xor(s, o); q += __shfl_xor(q, o); }
  float mu  = s * (1.0f / 128.0f);
  float var = q * (1.0f / 128.0f) - mu * mu;
  float rs  = rsqrtf(var + 1e-5f);
  float4 g4  = *(const float4*)(lg + lane * 4);
  float4 bb4 = *(const float4*)(lb + lane * 4);
  float4 r   = *(const float4*)(h + (size_t)n * D + lane * 4);
  float4 o4;
  o4.x = fmaxf((c.x - mu) * rs * g4.x + bb4.x, 0.0f) + r.x;
  o4.y = fmaxf((c.y - mu) * rs * g4.y + bb4.y, 0.0f) + r.y;
  o4.z = fmaxf((c.z - mu) * rs * g4.z + bb4.z, 0.0f) + r.z;
  o4.w = fmaxf((c.w - mu) * rs * g4.w + bb4.w, 0.0f) + r.w;
  *(float4*)(h + (size_t)n * D + lane * 4) = o4;
}

// -------------------- global mean pool (wave per node) --------------------
__global__ __launch_bounds__(256) void k_pool(
    const float* __restrict__ h, const int* __restrict__ batch,
    float* __restrict__ sums, float* __restrict__ cnt, int Nn) {
  int n    = blockIdx.x * 8 + (threadIdx.x >> 5);
  int lane = threadIdx.x & 31;
  if (n >= Nn) return;
  int b = batch[n];
  float4 v = *(const float4*)(h + (size_t)n * D + lane * 4);
  float* sp = sums + (size_t)b * D + lane * 4;
  atomicAdd(sp + 0, v.x);
  atomicAdd(sp + 1, v.y);
  atomicAdd(sp + 2, v.z);
  atomicAdd(sp + 3, v.w);
  if (lane == 0) atomicAdd(&cnt[b], 1.0f);
}

// -------------------- output MLP: relu(pooled@W1+b1)@W2+b2 --------------------
__global__ void k_mlp(const float* __restrict__ sums, const float* __restrict__ cnt,
                      const float* __restrict__ W1, const float* __restrict__ b1,
                      const float* __restrict__ W2, const float* __restrict__ b2,
                      float* __restrict__ out, int Gg) {
  int g = blockIdx.x * blockDim.x + threadIdx.x;
  if (g >= Gg) return;
  float inv = 1.0f / fmaxf(cnt[g], 1.0f);
  float o = b2[0];
  for (int j = 0; j < 64; ++j) {
    float a = 0.0f;
    for (int d = 0; d < 128; ++d) a += sums[(size_t)g * 128 + d] * W1[d * 64 + j];
    a = a * inv + b1[j];
    o += fmaxf(a, 0.0f) * W2[j];
  }
  out[g] = o;
}

// -------------------- driver --------------------
extern "C" void kernel_launch(void* const* d_in, const int* in_sizes, int n_in,
                              void* d_out, int out_size, void* d_ws, size_t ws_size,
                              hipStream_t stream) {
  (void)n_in; (void)ws_size;
  const float* x     = (const float*)d_in[0];
  const int*   ei    = (const int*)  d_in[1];
  const int*   batch = (const int*)  d_in[2];
  const float* W_in  = (const float*)d_in[3];
  const float* b_in  = (const float*)d_in[4];
  const float* linW  = (const float*)d_in[5];
  const float* asrc  = (const float*)d_in[6];
  const float* adst  = (const float*)d_in[7];
  const float* cbias = (const float*)d_in[8];
  const float* lng   = (const float*)d_in[9];
  const float* lnb   = (const float*)d_in[10];
  const float* W1    = (const float*)d_in[11];
  const float* b1    = (const float*)d_in[12];
  const float* W2    = (const float*)d_in[13];
  const float* b2    = (const float*)d_in[14];
  float* out = (float*)d_out;

  const int Nn   = in_sizes[0] / D;
  const int Ee   = in_sizes[1] / 2;
  const int Ll   = in_sizes[5] / (D * D);
  const int Gg   = out_size;
  const int Etot = Ee + Nn;

  // workspace layout
  float* ws   = (float*)d_ws;
  size_t off  = 0;
  float* hbuf = ws + off; off += (size_t)Nn * D;
  float* xh   = ws + off; off += (size_t)Nn * D;
  float* accb = ws + off; off += (size_t)Nn * D;
  float* ebuf = ws + off; off += (size_t)Etot * NH;
  float* esb  = ws + off; off += (size_t)Nn * NH;
  float* edb  = ws + off; off += (size_t)Nn * NH;
  float* mbuf = ws + off; off += (size_t)Nn * NH;
  float* zbuf = ws + off; off += (size_t)Nn * NH;
  float* sums = ws + off; off += (size_t)Gg * D;
  float* cntb = ws + off; off += (size_t)Gg;   // contiguous after sums

  const dim3 gemm_grid((Nn + 63) / 64);
  const int  edge_blks  = (Etot + 255) / 256;
  const int  edgew_blks = (Etot + 7) / 8;
  const int  nodew_blks = (Nn + 7) / 8;

  // input projection: h = x @ W_in + b_in
  k_gemm128<<<gemm_grid, 128, 0, stream>>>(x, W_in, b_in, hbuf, Nn);

  for (int l = 0; l < Ll; ++l) {
    k_gemm128<<<gemm_grid, 128, 0, stream>>>(hbuf, linW + (size_t)l * D * D, nullptr, xh, Nn);
    k_scores<<<(Nn * NH + 255) / 256, 256, 0, stream>>>(
        xh, asrc + (size_t)l * NH * HC, adst + (size_t)l * NH * HC, esb, edb, mbuf, zbuf, Nn);
    k_fill<<<((unsigned)((size_t)Nn * D + 255) / 256), 256, 0, stream>>>(accb, 0.0f, Nn * D);
    k_edge1<<<edge_blks, 256, 0, stream>>>(ei, esb, edb, ebuf, mbuf, Ee, Etot);
    k_edge2<<<edge_blks, 256, 0, stream>>>(ei, ebuf, mbuf, zbuf, Ee, Etot);
    k_edge3<<<edgew_blks, 256, 0, stream>>>(ei, ebuf, zbuf, xh, accb, Ee, Etot);
    k_finalize<<<nodew_blks, 256, 0, stream>>>(
        accb, cbias + (size_t)l * D, lng + (size_t)l * D, lnb + (size_t)l * D, hbuf, Nn);
  }

  k_fill<<<(Gg * D + Gg + 255) / 256, 256, 0, stream>>>(sums, 0.0f, Gg * D + Gg);
  k_pool<<<nodew_blks, 256, 0, stream>>>(hbuf, batch, sums, cntb, Nn);
  k_mlp<<<1, 64, 0, stream>>>(sums, cntb, W1, b1, W2, b2, out, Gg);
}
